// DWTEnhancedSTGCN_40776419508517
// MI455X (gfx1250) — matrix operand order repaired and checked by reading, compiled, not run on hardware
//
#include <hip/hip_runtime.h>
#include <hip/hip_bf16.h>

#define BB 8
#define TT 12
#define NN 10000
#define EE 160000

typedef __attribute__((ext_vector_type(16))) __bf16 v16bf;
typedef __attribute__((ext_vector_type(8)))  float  v8f;

// ---------------------------------------------------------------------------
// Weight prep: combined weights -> bf16 fragments in WMMA B-matrix layout.
// B (32x16, 16-bit) layout: lane holds column N = lane&15; lanes 0-15 hold
// K = kc*32 + 0..7 and 16..23 (2 per VGPR), lanes 16-31 hold K = kc*32 + 8..15 and 24..31.
// Fragment element i of lane L  <->  K = kc*32 + (i>=8?16:0) + (L>>4)*8 + (i&7).
// Storage: wfrag[mat][ ((kc*4 + j)*32 + lane)*16 + i ]   (j = 16-wide N tile)
// ---------------------------------------------------------------------------
__global__ void prep_weights_k(const float* Ws_h, const float* Wn_h,
                               const float* Ws_l, const float* Wn_l,
                               const float* Wc_l, const float* Whr,
                               const float* Wlr,
                               const float* b_h, const float* bhr,
                               const float* b_l, const float* blr,
                               __bf16* wfrag, float* bhp, float* blp) {
  int gid = blockIdx.x * 256 + threadIdx.x;
  const int totalW = 5 * 4096;
  if (gid < totalW) {
    int mat  = gid >> 12;
    int rem  = gid & 4095;
    int i    = rem & 15;
    int lane = (rem >> 4) & 31;
    int j    = (rem >> 9) & 3;
    int kc   = (rem >> 11) & 1;
    int k = kc * 32 + (i >= 8 ? 16 : 0) + (lane >> 4) * 8 + (i & 7);   // row of W (64x64)
    int n = j * 16 + (lane & 15);             // col of W
    int o = k * 64 + n;
    float v;
    if      (mat == 0) v = Ws_h[o] + 0.2f * Whr[o];  // combined high self
    else if (mat == 1) v = Wn_h[o];
    else if (mat == 2) v = Ws_l[o] + 0.2f * Wlr[o];  // combined low self
    else if (mat == 3) v = Wn_l[o];
    else               v = Wc_l[o];
    wfrag[gid] = (__bf16)v;
  } else if (gid < totalW + 64) {
    int d = gid - totalW;
    bhp[d] = b_h[d] + 0.2f * bhr[d];
  } else if (gid < totalW + 128) {
    int d = gid - totalW - 64;
    blp[d] = b_l[d] + 0.2f * blr[d];
  }
}

// ---------------------------------------------------------------------------
// Stage 1: ht/lt/resid = einsum('btn,td->bnd') (+ bias); resid uses 2*x.
// Block = 4 nodes x 64 feature-threads; x tile staged through LDS.
// ---------------------------------------------------------------------------
__global__ void stage1_k(const float* x,
                         const float* W_ht, const float* b_ht,
                         const float* W_lt, const float* b_lt,
                         const float* Wg,   const float* bg,
                         float* ht, float* lt, float* resid) {
  __shared__ float xs[4][TT];
  int tid = threadIdx.x;
  int ln  = tid >> 6;
  int d   = tid & 63;
  long p  = (long)blockIdx.x * 4 + ln;   // b*N + n
  int b   = (int)(p / NN);
  int n   = (int)(p % NN);
  if (d < TT) xs[ln][d] = x[((size_t)b * TT + d) * NN + n];
  __syncthreads();
  float hv = b_ht[d], lv = b_lt[d], rv = bg[d];
#pragma unroll
  for (int t = 0; t < TT; ++t) {
    float xv = xs[ln][t];
    hv = fmaf(xv, W_ht[t * 64 + d], hv);
    lv = fmaf(xv, W_lt[t * 64 + d], lv);
    rv = fmaf(2.0f * xv, Wg[t * 64 + d], rv);
  }
  size_t o = (size_t)p * 64 + d;
  ht[o] = hv; lt[o] = lv; resid[o] = rv;
}

// ---------------------------------------------------------------------------
// Degree counts (both edge lists in one pass).
// ---------------------------------------------------------------------------
__global__ void deg_k(const int* ei, const int* cei, float* deg, float* degc) {
  int e = blockIdx.x * 256 + threadIdx.x;
  if (e < EE) {
    atomicAdd(&deg[ei[EE + e]],  1.0f);
    atomicAdd(&degc[cei[EE + e]], 1.0f);
  }
}

// ---------------------------------------------------------------------------
// Scatter-add aggregation. Thread = (edge, feature); loop over batch.
// ---------------------------------------------------------------------------
__global__ void scatter2_k(const int* ei, const float* ht, const float* lt,
                           float* aggh, float* aggl) {
  int gid = blockIdx.x * 256 + threadIdx.x;   // e*64 + d
  int e = gid >> 6;
  int d = gid & 63;
  int s = ei[e];
  int t = ei[EE + e];
#pragma unroll
  for (int bb = 0; bb < BB; ++bb) {
    size_t so = ((size_t)bb * NN + s) * 64 + d;
    size_t to = ((size_t)bb * NN + t) * 64 + d;
    atomicAdd(&aggh[to], ht[so]);
    atomicAdd(&aggl[to], lt[so]);
  }
}

__global__ void scatter1_k(const int* cei, const float* lt, float* aggc) {
  int gid = blockIdx.x * 256 + threadIdx.x;
  int e = gid >> 6;
  int d = gid & 63;
  int s = cei[e];
  int t = cei[EE + e];
#pragma unroll
  for (int bb = 0; bb < BB; ++bb) {
    atomicAdd(&aggc[((size_t)bb * NN + t) * 64 + d],
              lt[((size_t)bb * NN + s) * 64 + d]);
  }
}

// ---------------------------------------------------------------------------
// Pack an A fragment (16x32 bf16). 16-bit A layout: lane supplies row
// M = lane&15; element i  <->  K = kc*32 + (i>=8?16:0) + (lane>>4)*8 + (i&7)
// => two contiguous 8-float runs at rowptr + kc*32 + grp*8 (+0, +16).
// ---------------------------------------------------------------------------
__device__ __forceinline__ v16bf a_frag(const float* p, float scale) {
  const float4* q = (const float4*)p;
  float4 A0 = q[0], A1 = q[1], A2 = q[4], A3 = q[5];
  float f[16] = {A0.x, A0.y, A0.z, A0.w, A1.x, A1.y, A1.z, A1.w,
                 A2.x, A2.y, A2.z, A2.w, A3.x, A3.y, A3.z, A3.w};
  v16bf out;
#pragma unroll
  for (int i = 0; i < 16; ++i) out[i] = (__bf16)(f[i] * scale);
  return out;
}

// ---------------------------------------------------------------------------
// Fused WMMA GEMM + LN + activations + attention fusion.
// One wave per (batch, 16-node tile). 40 v_wmma_f32_16x16x32_bf16 per wave.
// C/D layout: VGPR r of lane L holds (node = r + 8*(L>>4), d = j*16 + (L&15)).
// Per-node reductions (LN / logits) via shfl_xor over the 16-lane half-group.
// ---------------------------------------------------------------------------
__global__ void __launch_bounds__(256) fused_k(
    const float* ht, const float* lt, const float* resid,
    const float* aggh, const float* aggl, const float* aggc,
    const float* deg, const float* degc,
    const __bf16* wfrag, const float* bhp, const float* blp,
    const float* g_hn, const float* b_hn,
    const float* g_ln, const float* b_ln,
    const float* g_gn, const float* b_gn,
    const float* Wa, const float* ba,
    float* out_f, float* out_h, float* out_l) {
  int w    = blockIdx.x * 8 + (threadIdx.x >> 5);
  int b    = w / 625;
  int tile = w % 625;
  int n0   = tile * 16;
  int lane = threadIdx.x & 31;
  int grp  = lane >> 4;
  int l15  = lane & 15;

  __builtin_prefetch(wfrag + (size_t)lane * 64, 0, 0);  // global_prefetch_b8

  int rowA = n0 + l15;
  float invd  = 1.0f / fmaxf(deg[rowA],  1.0f);
  float invdc = 1.0f / fmaxf(degc[rowA], 1.0f);

  size_t rbase = ((size_t)b * NN + rowA) * 64;
  v16bf A_ht[2], A_ah[2], A_lt[2], A_al[2], A_ac[2];
#pragma unroll
  for (int kc = 0; kc < 2; ++kc) {
    int off = kc * 32 + grp * 8;
    A_ht[kc] = a_frag(ht   + rbase + off, 1.0f);
    A_ah[kc] = a_frag(aggh + rbase + off, invd);
    A_lt[kc] = a_frag(lt   + rbase + off, 1.0f);
    A_al[kc] = a_frag(aggl + rbase + off, invd);
    A_ac[kc] = a_frag(aggc + rbase + off, invdc);
  }

  const __bf16* W0 = wfrag;              // Ws_h + 0.2*Whr
  const __bf16* W1 = wfrag + 4096;       // Wn_h
  const __bf16* W2 = wfrag + 8192;       // Ws_l + 0.2*Wlr
  const __bf16* W3 = wfrag + 12288;      // Wn_l
  const __bf16* W4 = wfrag + 16384;      // Wc_l

  v8f zero = {0.f, 0.f, 0.f, 0.f, 0.f, 0.f, 0.f, 0.f};
  v8f acch[4], accl[4];
#pragma unroll
  for (int j = 0; j < 4; ++j) { acch[j] = zero; accl[j] = zero; }

#pragma unroll
  for (int j = 0; j < 4; ++j) {
#pragma unroll
    for (int kc = 0; kc < 2; ++kc) {
      int fo = ((kc * 4 + j) * 32 + lane) * 16;
      v16bf bw;
      bw = *(const v16bf*)(W0 + fo);
      acch[j] = __builtin_amdgcn_wmma_f32_16x16x32_bf16(
          false, A_ht[kc], false, bw, (short)0, acch[j], false, false);
      bw = *(const v16bf*)(W1 + fo);
      acch[j] = __builtin_amdgcn_wmma_f32_16x16x32_bf16(
          false, A_ah[kc], false, bw, (short)0, acch[j], false, false);
      bw = *(const v16bf*)(W2 + fo);
      accl[j] = __builtin_amdgcn_wmma_f32_16x16x32_bf16(
          false, A_lt[kc], false, bw, (short)0, accl[j], false, false);
      bw = *(const v16bf*)(W3 + fo);
      accl[j] = __builtin_amdgcn_wmma_f32_16x16x32_bf16(
          false, A_al[kc], false, bw, (short)0, accl[j], false, false);
      bw = *(const v16bf*)(W4 + fo);
      accl[j] = __builtin_amdgcn_wmma_f32_16x16x32_bf16(
          false, A_ac[kc], false, bw, (short)0, accl[j], false, false);
    }
  }

  // Per-(lane, j) small params
  float ghn[4], bhn[4], gln[4], bln[4], ggn[4], bgn[4], bhj[4], blj[4];
  float wah0[4], wah1[4], wal0[4], wal1[4];
#pragma unroll
  for (int j = 0; j < 4; ++j) {
    int d = j * 16 + l15;
    ghn[j] = g_hn[d]; bhn[j] = b_hn[d];
    gln[j] = g_ln[d]; bln[j] = b_ln[d];
    ggn[j] = g_gn[d]; bgn[j] = b_gn[d];
    bhj[j] = bhp[d];  blj[j] = blp[d];
    wah0[j] = Wa[d * 2 + 0];        wah1[j] = Wa[d * 2 + 1];
    wal0[j] = Wa[(64 + d) * 2 + 0]; wal1[j] = Wa[(64 + d) * 2 + 1];
  }
  float ba0 = ba[0], ba1 = ba[1];

  float hv[4][8], lv[4][8], rv[4][8];
#pragma unroll
  for (int j = 0; j < 4; ++j) {
#pragma unroll
    for (int r = 0; r < 8; ++r) {
      hv[j][r] = acch[j][r] + bhj[j];
      lv[j][r] = accl[j][r] + blj[j];
      int node = n0 + r + 8 * grp;
      rv[j][r] = resid[((size_t)b * NN + node) * 64 + j * 16 + l15];
    }
  }

#pragma unroll
  for (int r = 0; r < 8; ++r) {
    float sh = 0.f, qh = 0.f, sl = 0.f, ql = 0.f, sr = 0.f, qr = 0.f;
#pragma unroll
    for (int j = 0; j < 4; ++j) {
      sh += hv[j][r]; qh += hv[j][r] * hv[j][r];
      sl += lv[j][r]; ql += lv[j][r] * lv[j][r];
      sr += rv[j][r]; qr += rv[j][r] * rv[j][r];
    }
#pragma unroll
    for (int m = 1; m < 16; m <<= 1) {
      sh += __shfl_xor(sh, m, 32); qh += __shfl_xor(qh, m, 32);
      sl += __shfl_xor(sl, m, 32); ql += __shfl_xor(ql, m, 32);
      sr += __shfl_xor(sr, m, 32); qr += __shfl_xor(qr, m, 32);
    }
    const float inv64 = 1.0f / 64.0f;
    float mh = sh * inv64, vh = qh * inv64 - mh * mh;
    float ml = sl * inv64, vl = ql * inv64 - ml * ml;
    float mr = sr * inv64, vr = qr * inv64 - mr * mr;
    float rh = rsqrtf(vh + 1e-5f);
    float rl = rsqrtf(vl + 1e-5f);
    float rg = rsqrtf(vr + 1e-5f);

    float p0 = 0.f, p1 = 0.f;
#pragma unroll
    for (int j = 0; j < 4; ++j) {
      float h = (hv[j][r] - mh) * rh * ghn[j] + bhn[j];
      h = (h >= 0.f) ? h : 0.1f * h;                              // leaky relu
      float l = (lv[j][r] - ml) * rl * gln[j] + bln[j];
      l = 0.5f * l * (1.0f + erff(l * 0.70710678118654752f));     // exact gelu
      float g = (rv[j][r] - mr) * rg * ggn[j] + bgn[j];
      hv[j][r] = h; lv[j][r] = l; rv[j][r] = g;
      p0 += h * wah0[j] + l * wal0[j];
      p1 += h * wah1[j] + l * wal1[j];
    }
#pragma unroll
    for (int m = 1; m < 16; m <<= 1) {
      p0 += __shfl_xor(p0, m, 32);
      p1 += __shfl_xor(p1, m, 32);
    }
    float l0 = p0 + ba0, l1 = p1 + ba1;
    float a0 = 1.0f / (1.0f + expf(l1 - l0));   // softmax over 2 logits
    float a1 = 1.0f - a0;

    int node = n0 + r + 8 * grp;
#pragma unroll
    for (int j = 0; j < 4; ++j) {
      int d = j * 16 + l15;
      float h = hv[j][r], l = lv[j][r];
      float f = a0 * h + a1 * l + 0.3f * h + 0.3f * l + 0.1f * rv[j][r];
      size_t o = ((size_t)b * 64 + d) * NN + node;   // (B, OD, N) transposed
      out_f[o] = f; out_h[o] = h; out_l[o] = l;
    }
  }
}

// ---------------------------------------------------------------------------
extern "C" void kernel_launch(void* const* d_in, const int* in_sizes, int n_in,
                              void* d_out, int out_size, void* d_ws, size_t ws_size,
                              hipStream_t stream) {
  const float* x    = (const float*)d_in[0];
  const int*   ei   = (const int*)d_in[1];
  const int*   cei  = (const int*)d_in[2];
  const float* W_ht = (const float*)d_in[3];
  const float* b_ht = (const float*)d_in[4];
  const float* W_lt = (const float*)d_in[5];
  const float* b_lt = (const float*)d_in[6];
  const float* Ws_h = (const float*)d_in[7];
  const float* Wn_h = (const float*)d_in[8];
  const float* b_h  = (const float*)d_in[9];
  const float* Ws_l = (const float*)d_in[10];
  const float* Wn_l = (const float*)d_in[11];
  const float* Wc_l = (const float*)d_in[12];
  const float* b_l  = (const float*)d_in[13];
  const float* Whr  = (const float*)d_in[14];
  const float* bhr  = (const float*)d_in[15];
  const float* Wlr  = (const float*)d_in[16];
  const float* blr  = (const float*)d_in[17];
  const float* g_hn = (const float*)d_in[18];
  const float* b_hn = (const float*)d_in[19];
  const float* g_ln = (const float*)d_in[20];
  const float* b_ln = (const float*)d_in[21];
  const float* Wa   = (const float*)d_in[22];
  const float* ba   = (const float*)d_in[23];
  const float* Wg   = (const float*)d_in[24];
  const float* bg   = (const float*)d_in[25];
  const float* g_gn = (const float*)d_in[26];
  const float* b_gn = (const float*)d_in[27];

  float* ws = (float*)d_ws;
  const size_t F = (size_t)BB * NN * 64;       // 5,120,000 floats per feat map
  float* ht    = ws;
  float* lt    = ws + F;
  float* resid = ws + 2 * F;
  float* aggh  = ws + 3 * F;
  float* aggl  = ws + 4 * F;
  float* aggc  = ws + 5 * F;
  float* deg   = ws + 6 * F;
  float* degc  = deg + NN;
  float* bhp   = degc + NN;
  float* blp   = bhp + 64;
  __bf16* wfrag = (__bf16*)(blp + 64);

  // zero accumulators (agg maps + degrees) each launch
  hipMemsetAsync(aggh, 0, 3 * F * sizeof(float), stream);
  hipMemsetAsync(deg, 0, 2 * NN * sizeof(float), stream);

  prep_weights_k<<<(5 * 4096 + 128 + 255) / 256, 256, 0, stream>>>(
      Ws_h, Wn_h, Ws_l, Wn_l, Wc_l, Whr, Wlr, b_h, bhr, b_l, blr,
      wfrag, bhp, blp);

  stage1_k<<<(BB * NN) / 4, 256, 0, stream>>>(
      x, W_ht, b_ht, W_lt, b_lt, Wg, bg, ht, lt, resid);

  deg_k<<<(EE + 255) / 256, 256, 0, stream>>>(ei, cei, deg, degc);

  scatter2_k<<<(EE * 64) / 256, 256, 0, stream>>>(ei, ht, lt, aggh, aggl);
  scatter1_k<<<(EE * 64) / 256, 256, 0, stream>>>(cei, lt, aggc);

  float* out = (float*)d_out;
  fused_k<<<625, 256, 0, stream>>>(
      ht, lt, resid, aggh, aggl, aggc, deg, degc,
      wfrag, bhp, blp, g_hn, b_hn, g_ln, b_ln, g_gn, b_gn, Wa, ba,
      out, out + (size_t)BB * 64 * NN, out + 2 * (size_t)BB * 64 * NN);
}
